// MultiheadAttention_60868276519143
// MI455X (gfx1250) — matrix operand order repaired and checked by reading, not compiled, not run on hardware
//
#include <hip/hip_runtime.h>
#include <hip/hip_bf16.h>

// ---------------- problem constants ----------------
#define BATCH 2
#define SEQ   2048
#define DIM   1024
#define HEADS 16
#define HDIM  64   // DIM / HEADS

typedef _Float16 h16;
typedef h16   v16h __attribute__((ext_vector_type(16)));
typedef h16   v8h  __attribute__((ext_vector_type(8)));
typedef h16   v4h  __attribute__((ext_vector_type(4)));
typedef float v8f  __attribute__((ext_vector_type(8)));

#define SHUF16(lo, hi) __builtin_shufflevector((lo), (hi), 0,1,2,3,4,5,6,7,8,9,10,11,12,13,14,15)

static __device__ __forceinline__ v8f wmma_f16(v16h a, v16h b, v8f c) {
    return __builtin_amdgcn_wmma_f32_16x16x32_f16(false, a, false, b, (short)0, c, false, false);
}

// ---------------- CDNA5 async global->LDS copy (ASYNCcnt pipeline) ----------------
#if defined(__gfx1250__) && __has_builtin(__builtin_amdgcn_global_load_async_to_lds_b128)
#define HAVE_ASYNC 1
#else
#define HAVE_ASYNC 0
#endif

#if HAVE_ASYNC
typedef int i32x4 __attribute__((vector_size(16)));
typedef __attribute__((address_space(1))) i32x4 i32x4_g;   // global
typedef __attribute__((address_space(3))) i32x4 i32x4_l;   // LDS
#endif

static __device__ __forceinline__ void async_copy16(const h16* g, h16* l) {
#if HAVE_ASYNC
    __builtin_amdgcn_global_load_async_to_lds_b128(
        (i32x4_g*)g, (i32x4_l*)l, /*offset=*/0, /*cpol=*/0);
#else
    *(v8h*)l = *(const v8h*)g;   // synchronous fallback
#endif
}

static __device__ __forceinline__ void wait_async0() {
#if defined(__gfx1250__) && __has_builtin(__builtin_amdgcn_s_wait_asynccnt)
    __builtin_amdgcn_s_wait_asynccnt(0);
#elif HAVE_ASYNC
    asm volatile("s_wait_asynccnt 0" ::: "memory");
#endif
}

// =====================================================================
// Kernel 1: projection  Y = X @ W^T + b   (M=B*S=4096, N=DIM, K=DIM)
//   Output f16.  TRANSPOSE==0: Y[b,h,s,hd]  (Q,K)
//                TRANSPOSE==1: Y[b,h,hd,s]  (V, for the PV GEMM B-operand)
// Block: 128 threads = 4 waves; block tile 64x64; wave tile 32x32.
// =====================================================================
template <int TRANSPOSE>
__global__ __launch_bounds__(128)
void proj_kernel(const float* __restrict__ X, const float* __restrict__ W,
                 const float* __restrict__ bias, h16* __restrict__ Y)
{
    __shared__ __attribute__((aligned(16))) h16 Xs[64][40];
    __shared__ __attribute__((aligned(16))) h16 Ws[64][40];

    const int tid   = threadIdx.x;
    const int wave  = tid >> 5;
    const int lane  = tid & 31;
    const int lmod  = lane & 15;
    const int lhalf = lane >> 4;

    const int rowBase = blockIdx.y * 64;       // M tile base (B*S rows)
    const int colBase = blockIdx.x * 64;       // N tile base (DIM cols)
    const int wm = (wave >> 1) * 32;
    const int wn = (wave & 1) * 32;

    v8f c[2][2] = {};

    for (int k0 = 0; k0 < DIM; k0 += 32) {
        // ---- cooperative stage: 64x32 fp32 -> f16 LDS tiles ----
        {
            const int r  = tid >> 1;            // 0..63
            const int ch = (tid & 1) * 16;      // 0 or 16
            const float4* xs = (const float4*)(X + (size_t)(rowBase + r) * DIM + k0 + ch);
            const float4* ws = (const float4*)(W + (size_t)(colBase + r) * DIM + k0 + ch);
#pragma unroll
            for (int i = 0; i < 4; ++i) {
                float4 xv = xs[i];
                float4 wv = ws[i];
                v4h xp, wp;
                xp[0] = (h16)xv.x; xp[1] = (h16)xv.y; xp[2] = (h16)xv.z; xp[3] = (h16)xv.w;
                wp[0] = (h16)wv.x; wp[1] = (h16)wv.y; wp[2] = (h16)wv.z; wp[3] = (h16)wv.w;
                *(v4h*)&Xs[r][ch + i * 4] = xp;
                *(v4h*)&Ws[r][ch + i * 4] = wp;
            }
        }
        __syncthreads();

        // ---- A fragments (X rows), B fragments (W rows = output cols) ----
        v16h a[2], bf[2];
#pragma unroll
        for (int mi = 0; mi < 2; ++mi) {
            v8h lo = *(const v8h*)&Xs[wm + mi * 16 + lmod][lhalf * 8];
            v8h hi = *(const v8h*)&Xs[wm + mi * 16 + lmod][16 + lhalf * 8];
            a[mi] = SHUF16(lo, hi);
        }
#pragma unroll
        for (int ni = 0; ni < 2; ++ni) {
            v8h lo = *(const v8h*)&Ws[wn + ni * 16 + lmod][lhalf * 8];
            v8h hi = *(const v8h*)&Ws[wn + ni * 16 + lmod][16 + lhalf * 8];
            bf[ni] = SHUF16(lo, hi);
        }
#pragma unroll
        for (int mi = 0; mi < 2; ++mi)
#pragma unroll
            for (int ni = 0; ni < 2; ++ni)
                c[mi][ni] = wmma_f16(a[mi], bf[ni], c[mi][ni]);
        __syncthreads();
    }

    // ---- epilogue: +bias, branch-free scatter with compile-time stride ----
#pragma unroll
    for (int ni = 0; ni < 2; ++ni) {
        const int gn = colBase + wn + ni * 16 + lmod;   // 0..DIM-1
        const float bv = bias[gn];
        const int hh = gn >> 6;          // head
        const int hd = gn & 63;          // within head
#pragma unroll
        for (int mi = 0; mi < 2; ++mi) {
            const int gm0 = rowBase + wm + mi * 16 + lhalf * 8;  // 8 consecutive rows
            const int b   = gm0 >> 11;        // / SEQ (tile never crosses batch)
            const int s0  = gm0 & 2047;       // % SEQ
            h16* dst;
            if (TRANSPOSE)
                dst = Y + ((size_t)((b * HEADS + hh) * HDIM + hd)) * SEQ + s0;
            else
                dst = Y + ((size_t)((b * HEADS + hh) * SEQ + s0)) * HDIM + hd;
            const size_t stride = TRANSPOSE ? 1 : HDIM;
#pragma unroll
            for (int r = 0; r < 8; ++r)
                dst[r * stride] = (h16)(c[mi][ni][r] + bv);
        }
    }
}

// =====================================================================
// Kernel 2: flash attention over f16 Q[b,h,s,hd], K[b,h,s,hd], Vt[b,h,hd,s]
// Block: 256 threads = 8 waves; block = 128 query rows; wave = 16 rows.
// Inner loop: 32-key tiles, double-buffered async global->LDS K/V staging.
// =====================================================================
__global__ __launch_bounds__(256)
void attn_kernel(const h16* __restrict__ Qh, const h16* __restrict__ Kh,
                 const h16* __restrict__ Vt, const float* __restrict__ amask,
                 const unsigned char* __restrict__ kpad, float* __restrict__ out)
{
    __shared__ __attribute__((aligned(16))) h16 Ks[2][32][72];    // [buf][key][hd]
    __shared__ __attribute__((aligned(16))) h16 Vs[2][64][40];    // [buf][hd][key]
    __shared__ __attribute__((aligned(16))) h16 Ps[8][16][40];    // per-wave P tile

    const int tid   = threadIdx.x;
    const int wave  = tid >> 5;
    const int lane  = tid & 31;
    const int lmod  = lane & 15;
    const int lhalf = lane >> 4;

    const int bh = blockIdx.y;          // b*HEADS + h
    const int b  = bh >> 4;
    const int h  = bh & 15;
    const int qBase = blockIdx.x * 128 + wave * 16;

    const float NEG_INF = -__builtin_inff();

    // per-thread staging coordinates (one b128 per thread per tile)
    const int kr = tid >> 3, ch = (tid & 7) * 8;   // K tile: 32 x 64
    const int vd = tid >> 2, vc = (tid & 3) * 8;   // V tile: 64 x 32
    const h16* Kg = Kh + ((size_t)bh * SEQ + kr) * HDIM + ch;   // += kb*HDIM
    const h16* Vg = Vt + ((size_t)bh * HDIM + vd) * SEQ + vc;   // += kb

    // ---- Q A-fragments, loaded once (two K-steps over Hd=64) ----
    v16h qa[2];
    {
        const h16* qrow = Qh + ((size_t)bh * SEQ + qBase + lmod) * HDIM;
#pragma unroll
        for (int st = 0; st < 2; ++st) {
            v8h lo = *(const v8h*)(qrow + st * 32 + lhalf * 8);
            v8h hi = *(const v8h*)(qrow + st * 32 + 16 + lhalf * 8);
            qa[st] = SHUF16(lo, hi);
        }
    }

    v8f o[4] = {};
    float rmax[8], rsum[8];
#pragma unroll
    for (int r = 0; r < 8; ++r) { rmax[r] = NEG_INF; rsum[r] = 0.0f; }

    // prologue: stage tile 0 into buffer 0
    async_copy16(Kg, &Ks[0][kr][ch]);
    async_copy16(Vg, &Vs[0][vd][vc]);

    const int NT = SEQ / 32;
    for (int it = 0; it < NT; ++it) {
        wait_async0();        // my async copies into buf it&1 have landed
        __syncthreads();      // everyone's landed; everyone done reading buf (it+1)&1

        // kick off next tile into the other buffer while we compute
        if (it + 1 < NT) {
            const int kb1 = (it + 1) * 32;
            async_copy16(Kg + (size_t)kb1 * HDIM, &Ks[(it + 1) & 1][kr][ch]);
            async_copy16(Vg + kb1,                &Vs[(it + 1) & 1][vd][vc]);
        }

        const int buf = it & 1;
        const int kb  = it * 32;

        // ---- scores: S = (Q K^T) / 8 + attn_mask + pad_mask ----
        float p[2][8];
#pragma unroll
        for (int nt = 0; nt < 2; ++nt) {
            v16h kf[2];
#pragma unroll
            for (int st = 0; st < 2; ++st) {
                v8h lo = *(const v8h*)&Ks[buf][nt * 16 + lmod][st * 32 + lhalf * 8];
                v8h hi = *(const v8h*)&Ks[buf][nt * 16 + lmod][st * 32 + 16 + lhalf * 8];
                kf[st] = SHUF16(lo, hi);
            }
            v8f acc = {};
            acc = wmma_f16(qa[0], kf[0], acc);
            acc = wmma_f16(qa[1], kf[1], acc);

            const int kcol = kb + nt * 16 + lmod;
            const float padv = kpad[b * SEQ + kcol] ? NEG_INF : 0.0f;
#pragma unroll
            for (int r = 0; r < 8; ++r) {
                const int qm = qBase + lhalf * 8 + r;
                p[nt][r] = acc[r] * 0.125f + amask[(size_t)qm * SEQ + kcol] + padv;
            }
        }

        // ---- online softmax update (row = across 16 lanes of half-wave) ----
#pragma unroll
        for (int r = 0; r < 8; ++r) {
            float m2 = fmaxf(p[0][r], p[1][r]);
            m2 = fmaxf(m2, __shfl_xor(m2, 1));
            m2 = fmaxf(m2, __shfl_xor(m2, 2));
            m2 = fmaxf(m2, __shfl_xor(m2, 4));
            m2 = fmaxf(m2, __shfl_xor(m2, 8));
            const float nm = fmaxf(rmax[r], m2);
            const float mu = (nm == NEG_INF) ? 0.0f : nm;  // all-masked guard
            const float alpha = __expf(rmax[r] - mu);
            const float p0 = __expf(p[0][r] - mu);
            const float p1 = __expf(p[1][r] - mu);
            p[0][r] = p0; p[1][r] = p1;
            float ls = p0 + p1;
            ls += __shfl_xor(ls, 1);
            ls += __shfl_xor(ls, 2);
            ls += __shfl_xor(ls, 4);
            ls += __shfl_xor(ls, 8);
            rsum[r] = rsum[r] * alpha + ls;
            rmax[r] = nm;
#pragma unroll
            for (int nt = 0; nt < 4; ++nt) o[nt][r] *= alpha;
        }

        // ---- round-trip P through per-wave LDS to get A-layout ----
#pragma unroll
        for (int nt = 0; nt < 2; ++nt)
#pragma unroll
            for (int r = 0; r < 8; ++r)
                Ps[wave][lhalf * 8 + r][nt * 16 + lmod] = (h16)p[nt][r];
        asm volatile("" ::: "memory");   // same-wave LDS ops are in-order (ISA §7.3)
        v16h pa;
        {
            v8h lo = *(const v8h*)&Ps[wave][lmod][lhalf * 8];
            v8h hi = *(const v8h*)&Ps[wave][lmod][16 + lhalf * 8];
            pa = SHUF16(lo, hi);
        }

        // ---- O += P V  (N = 64 -> four 16-wide tiles) ----
#pragma unroll
        for (int nt = 0; nt < 4; ++nt) {
            v8h lo = *(const v8h*)&Vs[buf][nt * 16 + lmod][lhalf * 8];
            v8h hi = *(const v8h*)&Vs[buf][nt * 16 + lmod][16 + lhalf * 8];
            v16h vf = SHUF16(lo, hi);
            o[nt] = wmma_f16(pa, vf, o[nt]);
        }
    }

    // ---- normalize and write f32 output [B, S, H*Hd] ----
#pragma unroll
    for (int r = 0; r < 8; ++r) {
        const float inv = 1.0f / rsum[r];
        const int qm = qBase + lhalf * 8 + r;
        float* orow = out + ((size_t)(b * SEQ + qm)) * DIM + h * HDIM;
#pragma unroll
        for (int nt = 0; nt < 4; ++nt)
            orow[nt * 16 + lmod] = o[nt][r] * inv;
    }
}

// =====================================================================
extern "C" void kernel_launch(void* const* d_in, const int* in_sizes, int n_in,
                              void* d_out, int out_size, void* d_ws, size_t ws_size,
                              hipStream_t stream)
{
    const float* query = (const float*)d_in[0];
    const float* key   = (const float*)d_in[1];
    const float* value = (const float*)d_in[2];
    const float* amask = (const float*)d_in[3];
    const unsigned char* kpad = (const unsigned char*)d_in[4];
    const float* Wq = (const float*)d_in[5];
    const float* bq = (const float*)d_in[6];
    const float* Wk = (const float*)d_in[7];
    const float* bk = (const float*)d_in[8];
    const float* Wv = (const float*)d_in[9];
    const float* bv = (const float*)d_in[10];
    float* out = (float*)d_out;

    const size_t per = (size_t)BATCH * HEADS * SEQ * HDIM;   // 4M halves each
    h16* Qh = (h16*)d_ws;
    h16* Kh = Qh + per;
    h16* Vt = Kh + per;

    dim3 pgrid(DIM / 64, (BATCH * SEQ) / 64);   // (16, 64)
    proj_kernel<0><<<pgrid, 128, 0, stream>>>(query, Wq, bq, Qh);
    proj_kernel<0><<<pgrid, 128, 0, stream>>>(key,   Wk, bk, Kh);
    proj_kernel<1><<<pgrid, 128, 0, stream>>>(value, Wv, bv, Vt);

    dim3 agrid(SEQ / 128, BATCH * HEADS);       // (16, 32)
    attn_kernel<<<agrid, 256, 0, stream>>>(Qh, Kh, Vt, amask, kpad, out);
}
